// DGI_7267084665520
// MI455X (gfx1250) — matrix operand order repaired and checked by reading, not compiled, not run on hardware
//
#include <hip/hip_runtime.h>

#define NN 8192
#define FIN 256
#define HH 64
#define KT (NN / 32)              // 256 k-tiles of 32
#define KSPLIT 4                  // k-range split for occupancy
#define KT_PER (KT / KSPLIT)      // 64 k-tiles per wave
#define FTSB_U32 (KT * 1024)      // 262144 dwords per sequence (packed B operand)
#define FTSB_U4  (FTSB_U32 / 4)   // 65536 uint4 per sequence

typedef __attribute__((ext_vector_type(16))) __bf16 v16bf;
typedef __attribute__((ext_vector_type(8)))  float  v8f;
typedef __attribute__((ext_vector_type(4)))  float  f4v;

union BfVec { unsigned int u[8]; uint4 q[2]; v16bf v; };
union AReg  { f4v v4[4]; float f[16]; };

__device__ __forceinline__ unsigned short f2bf(float f) {
    unsigned int u = __float_as_uint(f);
    u += 0x7FFFu + ((u >> 16) & 1u);          // round-to-nearest-even
    return (unsigned short)(u >> 16);
}
__device__ __forceinline__ unsigned int pk2bf(float lo, float hi) {
    return (unsigned int)f2bf(lo) | ((unsigned int)f2bf(hi) << 16);
}

// gfx1250 async DMA: global -> LDS, 16B per lane, tracked by ASYNCcnt.
// LDS address operand = wave-relative byte offset (= low 32 bits of the
// flat address of a __shared__ object, per ISA 10.2 LDS aperture mapping).
__device__ __forceinline__ unsigned ldsOff(const void* p) {
    return (unsigned)(unsigned long long)p;
}
__device__ __forceinline__ void asyncCopyB128(unsigned lds_byte_off, const void* gsrc) {
    asm volatile("global_load_async_to_lds_b128 %0, %1, off"
                 :: "v"(lds_byte_off), "v"((unsigned long long)gsrc)
                 : "memory");
}
__device__ __forceinline__ void waitAsync0() {
    asm volatile("s_wait_asynccnt 0x0" ::: "memory");
}

// ---------------------------------------------------------------------------
// Kernel 1: fts_j = seq_j @ fc_w^T, stored bf16 pre-swizzled into the WMMA
// B-operand layout. Per (kb) tile plane (256 uint4): uint4 index =
//   hb*64 + q*32 + lane   (q = v>>2), dword vi = v&3, dword holds elements s=2v,2v+1
// (lanes 0-15: K 0..7,16..23; lanes 16-31: K 8..15,24..31), column = hb*16 + lane%16.
// Also zero-initializes cpart and the hbuf accumulation buffer.
// ---------------------------------------------------------------------------
__global__ __launch_bounds__(256) void dgi_fts_pack(
    const float* __restrict__ seq1, const float* __restrict__ seq2,
    const float* __restrict__ fcw, unsigned short* __restrict__ ftsB,
    float* __restrict__ hbuf, float* __restrict__ cpart)
{
    if (blockIdx.x == 0 && threadIdx.x < 64) cpart[threadIdx.x] = 0.0f;

    int gid = blockIdx.x * 256 + threadIdx.x;     // NN*HH threads total
    hbuf[gid] = 0.0f;                             // zero h accumulation (j=0)
    hbuf[gid + NN * HH] = 0.0f;                   // zero h accumulation (j=1)

    int m = gid >> 6;
    int h = gid & 63;
    const float4* s1 = (const float4*)(seq1 + (size_t)m * FIN);
    const float4* s2 = (const float4*)(seq2 + (size_t)m * FIN);
    const float4* w  = (const float4*)(fcw  + (size_t)h * FIN);
    float a1 = 0.f, a2 = 0.f;
    #pragma unroll 8
    for (int i = 0; i < FIN / 4; ++i) {
        float4 wv = w[i], x = s1[i], y = s2[i];
        a1 = fmaf(wv.x, x.x, fmaf(wv.y, x.y, fmaf(wv.z, x.z, fmaf(wv.w, x.w, a1))));
        a2 = fmaf(wv.x, y.x, fmaf(wv.y, y.y, fmaf(wv.z, y.z, fmaf(wv.w, y.w, a2))));
    }
    int kb = m >> 5, kr = m & 31;
    int v = ((kr & 7) | ((kr & 16) >> 1)) >> 1, half = kr & 1;
    int lane = ((kr & 8) ? 16 : 0) + (h & 15);
    int hb = h >> 4;
    size_t dword = (size_t)kb * 1024 + hb * 256 + (v >> 2) * 128 + lane * 4 + (v & 3);
    ftsB[dword * 2 + half]                        = f2bf(a1);
    ftsB[((size_t)FTSB_U32 + dword) * 2 + half]   = f2bf(a2);
}

// ---------------------------------------------------------------------------
// Kernel 2: fused partial out_j = adj @ fts_j for j=1,2, ONE NT-streamed pass
// of adj (256MB). Grid = 128 M-groups x KSPLIT k-ranges, 4 waves/block.
// B slice for the next iteration is DMA'd global->LDS with
// global_load_async_to_lds_b128 (ASYNCcnt) -- zero VGPR staging, no spills.
// A tile (16x32 f32 adj) NT-loaded per A-matrix lane layout, cvt to bf16
// in-reg; 8 WMMAs per iteration. Partials combined into hbuf via f32 atomics.
// ---------------------------------------------------------------------------
__global__ __launch_bounds__(128) void dgi_gcn_wmma(
    const float* __restrict__ adj,
    const uint4* __restrict__ ftsB,       // [2][KT][256] uint4
    float* __restrict__ hbuf)             // [2*NN][HH] accumulators
{
    __shared__ uint4 smem[2][2][256];     // 16 KB double-buffered B slice

    const int tid = threadIdx.x;
    const int lane = tid & 31;
    const int wave = tid >> 5;
    const int laneRow = lane & 15;
    const int laneSel = lane >> 4;
    const int ks   = blockIdx.x & (KSPLIT - 1);
    const int mgrp = blockIdx.x >> 2;
    const int n0 = (mgrp * 4 + wave) * 16;
    const int kb0 = ks * KT_PER;
    const int kb1 = kb0 + KT_PER;

    // A-layout: lanes 0-15 hold K {0..7,16..23}; lanes 16-31 hold K {8..15,24..31}
    const float* arow = adj + (size_t)(n0 + laneRow) * NN + laneSel * 8;

    v8f acc[2][4];
    #pragma unroll
    for (int j = 0; j < 2; ++j)
        #pragma unroll
        for (int hb = 0; hb < 4; ++hb) acc[j][hb] = (v8f)0.f;

    // ---- prologue: async-stage tile kb0, prefetch A(kb0) into regs ----
    #pragma unroll
    for (int j = 0; j < 2; ++j) {
        const uint4* gp = ftsB + (size_t)j * FTSB_U4 + (size_t)kb0 * 256;
        asyncCopyB128(ldsOff(&smem[0][j][tid]),       gp + tid);
        asyncCopyB128(ldsOff(&smem[0][j][tid + 128]), gp + tid + 128);
    }
    AReg areg;
    {
        const f4v* ap = (const f4v*)(arow + (size_t)kb0 * 32);
        areg.v4[0] = __builtin_nontemporal_load(ap + 0);
        areg.v4[1] = __builtin_nontemporal_load(ap + 1);
        areg.v4[2] = __builtin_nontemporal_load(ap + 4);
        areg.v4[3] = __builtin_nontemporal_load(ap + 5);
    }
    waitAsync0();
    __syncthreads();

    int buf = 0;
    for (int kb = kb0; kb < kb1; ++kb) {
        const bool more = (kb + 1 < kb1);
        AReg anext;
        if (more) {
            // async DMA next B slice straight into the other LDS buffer
            #pragma unroll
            for (int j = 0; j < 2; ++j) {
                const uint4* gp = ftsB + (size_t)j * FTSB_U4 + (size_t)(kb + 1) * 256;
                asyncCopyB128(ldsOff(&smem[buf ^ 1][j][tid]),       gp + tid);
                asyncCopyB128(ldsOff(&smem[buf ^ 1][j][tid + 128]), gp + tid + 128);
            }
            // prefetch next A tile into regs (NT: don't pollute L2)
            const f4v* ap = (const f4v*)(arow + (size_t)(kb + 1) * 32);
            anext.v4[0] = __builtin_nontemporal_load(ap + 0);
            anext.v4[1] = __builtin_nontemporal_load(ap + 1);
            anext.v4[2] = __builtin_nontemporal_load(ap + 4);
            anext.v4[3] = __builtin_nontemporal_load(ap + 5);
        }

        // convert current A tile f32 -> packed bf16 (RNE); areg dies here
        BfVec a;
        #pragma unroll
        for (int i = 0; i < 8; ++i) a.u[i] = pk2bf(areg.f[2 * i], areg.f[2 * i + 1]);

        // 8 WMMAs sharing one A operand (B from LDS, 2 x b128 per operand)
        #pragma unroll
        for (int j = 0; j < 2; ++j) {
            BfVec b0, b1, b2, b3;
            b0.q[0] = smem[buf][j][0 * 64 + lane];
            b0.q[1] = smem[buf][j][0 * 64 + 32 + lane];
            b1.q[0] = smem[buf][j][1 * 64 + lane];
            b1.q[1] = smem[buf][j][1 * 64 + 32 + lane];
            b2.q[0] = smem[buf][j][2 * 64 + lane];
            b2.q[1] = smem[buf][j][2 * 64 + 32 + lane];
            b3.q[0] = smem[buf][j][3 * 64 + lane];
            b3.q[1] = smem[buf][j][3 * 64 + 32 + lane];
            acc[j][0] = __builtin_amdgcn_wmma_f32_16x16x32_bf16(
                false, a.v, false, b0.v, (short)0, acc[j][0], false, false);
            acc[j][1] = __builtin_amdgcn_wmma_f32_16x16x32_bf16(
                false, a.v, false, b1.v, (short)0, acc[j][1], false, false);
            acc[j][2] = __builtin_amdgcn_wmma_f32_16x16x32_bf16(
                false, a.v, false, b2.v, (short)0, acc[j][2], false, false);
            acc[j][3] = __builtin_amdgcn_wmma_f32_16x16x32_bf16(
                false, a.v, false, b3.v, (short)0, acc[j][3], false, false);
        }

        if (more) areg = anext;
        waitAsync0();          // this wave's async copies into buf^1 are done
        __syncthreads();       // all waves' copies published
        buf ^= 1;
    }

    // ---- combine partial accumulators into hbuf (distinct-address atomics) ----
    #pragma unroll
    for (int r = 0; r < 8; ++r) {
        int row = n0 + r + 8 * laneSel;   // C/D layout: VGPR r -> M = r + 8*(lane/16)
        #pragma unroll
        for (int hb = 0; hb < 4; ++hb) {
            int col = hb * 16 + laneRow;
            unsafeAtomicAdd(&hbuf[(size_t)row * HH + col],        acc[0][hb][r]);
            unsafeAtomicAdd(&hbuf[(size_t)(NN + row) * HH + col], acc[1][hb][r]);
        }
    }
}

// ---------------------------------------------------------------------------
// Kernel 2b: bias + PReLU in-place on hbuf; masked column-sum of h1 into
// cpart via two-stage (LDS then global) reduction. Block = 64 rows x 64 cols.
// ---------------------------------------------------------------------------
__global__ __launch_bounds__(256) void dgi_epilogue(
    float* __restrict__ hbuf,
    const float* __restrict__ bias,
    const float* __restrict__ alphap,
    const float* __restrict__ msk,
    float* __restrict__ cpart)
{
    __shared__ float sred[64];
    const int b = blockIdx.x;            // 0..255
    const int j = b >> 7;
    const int rblk = b & 127;
    const int t = threadIdx.x;
    const int col = t & 63;
    const int rsub = t >> 6;             // 0..3
    const float alpha = *alphap;
    const float bz = bias[col];

    float csum = 0.f;
    #pragma unroll 4
    for (int i = 0; i < 16; ++i) {
        int row = rblk * 64 + rsub + i * 4;
        size_t idx = ((size_t)(j * NN + row)) * HH + col;
        float v = hbuf[idx] + bz;
        v = v > 0.f ? v : alpha * v;
        hbuf[idx] = v;
        if (j == 0) csum = fmaf(v, msk[row], csum);
    }
    if (j == 0) {
        if (t < 64) sred[t] = 0.f;
        __syncthreads();
        atomicAdd(&sred[col], csum);                 // ds_add_f32
        __syncthreads();
        if (t < 64) unsafeAtomicAdd(&cpart[t], sred[t]);
    }
}

// ---------------------------------------------------------------------------
// Kernel 3: c = sigmoid(cpart / sum(msk)); wc = disc_w @ c
// ---------------------------------------------------------------------------
__global__ void dgi_cw(const float* __restrict__ cpart,
                       const float* __restrict__ msk,
                       const float* __restrict__ discw,
                       float* __restrict__ wc)
{
    __shared__ float cs[64];
    __shared__ float red[64];
    int t = threadIdx.x;
    float p = 0.f;
    for (int i = t; i < NN; i += 64) p += msk[i];
    red[t] = p; __syncthreads();
    for (int s = 32; s > 0; s >>= 1) {
        if (t < s) red[t] += red[t + s];
        __syncthreads();
    }
    float cv = cpart[t] / red[0];
    cs[t] = 1.f / (1.f + __expf(-cv));
    __syncthreads();
    const float* wrow = discw + t * 64;
    float acc = 0.f;
    #pragma unroll 8
    for (int k = 0; k < 64; ++k) acc = fmaf(wrow[k], cs[k], acc);
    wc[t] = acc;
}

// ---------------------------------------------------------------------------
// Kernel 4: sc[j,n] = h_j[n,:] . wc + disc_b  ->  d_out[0:2N]
// ---------------------------------------------------------------------------
__global__ __launch_bounds__(256) void dgi_score(
    const float* __restrict__ hbuf,   // [2*NN][HH]
    const float* __restrict__ wc,
    const float* __restrict__ discb,
    float* __restrict__ out)
{
    __shared__ __align__(16) float w[64];
    int t = threadIdx.x;
    if (t < 64) w[t] = wc[t];
    __syncthreads();
    int idx = blockIdx.x * 256 + t;                 // 0 .. 2*NN-1
    const float4* hp = (const float4*)(hbuf + (size_t)idx * HH);
    const float4* wp = (const float4*)w;
    float acc = 0.f;
    #pragma unroll
    for (int i = 0; i < 16; ++i) {
        float4 hv = hp[i], wv = wp[i];
        acc = fmaf(hv.x, wv.x, fmaf(hv.y, wv.y, fmaf(hv.z, wv.z, fmaf(hv.w, wv.w, acc))));
    }
    out[idx] = acc + discb[0];
}

// ---------------------------------------------------------------------------
extern "C" void kernel_launch(void* const* d_in, const int* in_sizes, int n_in,
                              void* d_out, int out_size, void* d_ws, size_t ws_size,
                              hipStream_t stream) {
    const float* seq1  = (const float*)d_in[0];
    const float* seq2  = (const float*)d_in[1];
    const float* adj   = (const float*)d_in[2];
    const float* msk   = (const float*)d_in[3];
    const float* fcw   = (const float*)d_in[4];
    const float* bias  = (const float*)d_in[5];
    const float* alpha = (const float*)d_in[6];
    const float* discw = (const float*)d_in[7];
    const float* discb = (const float*)d_in[8];
    float* out = (float*)d_out;

    char* ws = (char*)d_ws;
    unsigned short* ftsB16 = (unsigned short*)ws;                  // 2 MB packed bf16 B
    const uint4*    ftsB4  = (const uint4*)ws;
    float* hbuf  = (float*)(ws + (size_t)2 * FTSB_U32 * 4);        // 4 MB h1,h2
    float* cpart = (float*)(ws + (size_t)2 * FTSB_U32 * 4 + (size_t)2 * NN * HH * 4);
    float* wc    = cpart + 64;

    dgi_fts_pack<<<(NN * HH) / 256, 256, 0, stream>>>(seq1, seq2, fcw, ftsB16, hbuf, cpart);
    dgi_gcn_wmma<<<(NN / 64) * KSPLIT, 128, 0, stream>>>(adj, ftsB4, hbuf);
    dgi_epilogue<<<256, 256, 0, stream>>>(hbuf, bias, alpha, msk, cpart);
    dgi_cw<<<1, 64, 0, stream>>>(cpart, msk, discw, wc);
    dgi_score<<<(2 * NN) / 256, 256, 0, stream>>>(hbuf, wc, discb, out);
}
